// AdaptiveBCEWithLogitsLoss_72885595013388
// MI455X (gfx1250) — compile-verified, matrix operand order checked by reading, not run on hardware
//
#include <hip/hip_runtime.h>
#include <hip/hip_bf16.h>
#include <math.h>

#define IN_F   768
#define BATCH  256
#define SHORTN 2000

typedef __attribute__((ext_vector_type(16))) __bf16    v16bf;
typedef __attribute__((ext_vector_type(8)))  float     v8f;
typedef __attribute__((ext_vector_type(8)))  unsigned  v8u;

// ---- workspace layout (float offsets) ----
enum : size_t {
  OFF_H0       = 0,                    // 256*384
  OFF_H1       = OFF_H0 + 256*384,     // 256*192
  OFF_H2       = OFF_H1 + 256*192,     // 256*96
  OFF_ROOT_RAW = OFF_H2 + 256*96,      // 3*256
  OFF_ROOT_SIG = OFF_ROOT_RAW + 768,   // 3*256
  OFF_ACT      = OFF_ROOT_SIG + 768,   // 3*256
  OFF_NUML     = OFF_ACT + 768,        // 256
  OFF_HEADSUM  = OFF_NUML + 256,       // 256
  OFF_CLUSSUM  = OFF_HEADSUM + 256,    // 256
};

__device__ __forceinline__ unsigned pack2bf(float lo, float hi) {
  unsigned short a = __builtin_bit_cast(unsigned short, (__bf16)lo);
  unsigned short b = __builtin_bit_cast(unsigned short, (__bf16)hi);
  return (unsigned)a | ((unsigned)b << 16);
}

// K index within a 16x32 fragment for (vgpr v, half): ISA 7.12.2 16-bit A layout.
__device__ __forceinline__ int frag_bk(int v, int half) {
  return ((v < 4) ? (v * 2) : (16 + (v - 4) * 2)) + half * 8;
}

// Load a 16x32 fp32 tile as a bf16 WMMA fragment (native converts).
__device__ __forceinline__ v16bf load_frag(const float* __restrict__ base, int ld,
                                           int row0, int k0, int lane) {
  v16bf f;
  int half = lane >> 4;
  const float* p = base + (size_t)(row0 + (lane & 15)) * (size_t)ld + k0;
  const float2* p2 = reinterpret_cast<const float2*>(p);
#pragma unroll
  for (int v = 0; v < 8; ++v) {
    float2 t = p2[frag_bk(v, half) >> 1];
    f[2 * v]     = (__bf16)t.x;
    f[2 * v + 1] = (__bf16)t.y;
  }
  return f;
}

__device__ __forceinline__ float bce0_logit(float t) {   // BCEWithLogits, y = 0
  return fmaxf(t, 0.f) + log1pf(expf(-fabsf(t)));
}

// ---------------- prep: active flags, num_loss, root logits, zero accums ----
__global__ void k_prep_abce(const float* __restrict__ x, const float* __restrict__ hW,
                            const int* __restrict__ tgt, float* __restrict__ ws) {
  int b = threadIdx.x;                       // 256 threads, one per row
  ws[OFF_HEADSUM + b] = 0.f;
  ws[OFF_CLUSSUM + b] = 0.f;
  const int* tg = tgt + b * 20;
  int a0 = 0, a1 = 0, a2 = 0;
  for (int l = 0; l < 20; ++l) {
    int t = tg[l];
    a0 |= (t >= 2000 && t < 12000);
    a1 |= (t >= 12000 && t < 40000);
    a2 |= (t >= 40000);
  }
  ws[OFF_ACT + 0 * 256 + b] = (float)a0;
  ws[OFF_ACT + 1 * 256 + b] = (float)a1;
  ws[OFF_ACT + 2 * 256 + b] = (float)a2;
  float nl = (float)SHORTN;
  nl += a0 ? 10000.f : 1.f;
  nl += a1 ? 28000.f : 1.f;
  nl += a2 ? 60000.f : 1.f;
  ws[OFF_NUML + b] = nl;
  const float* xr = x + (size_t)b * IN_F;
  for (int i = 0; i < 3; ++i) {
    const float* wr = hW + (size_t)(SHORTN + i) * IN_F;
    float d = 0.f;
    for (int k = 0; k < IN_F; ++k) d = fmaf(xr[k], wr[k], d);
    ws[OFF_ROOT_RAW + i * 256 + b] = d;
    ws[OFF_ROOT_SIG + i * 256 + b] = 1.f / (1.f + expf(-d));
  }
}

// ---------------- small WMMA GEMM: out[M=256,N] = A[256,K] * B[N,K]^T --------
__global__ void k_gemm16_abce(const float* __restrict__ A, const float* __restrict__ Bm,
                              float* __restrict__ out, int N, int K) {
  int lane = threadIdx.x & 31, w = threadIdx.x >> 5;
  int n0 = blockIdx.x * 16;
  int m0 = (blockIdx.y * 8 + w) * 16;
  v8f c = {0.f, 0.f, 0.f, 0.f, 0.f, 0.f, 0.f, 0.f};
  for (int k0 = 0; k0 < K; k0 += 32) {
    v16bf a = load_frag(A,  K, m0, k0, lane);
    v16bf b = load_frag(Bm, K, n0, k0, lane);
    c = __builtin_amdgcn_wmma_f32_16x16x32_bf16(false, a, false, b, (short)0, c, false, false);
  }
  int half = lane >> 4, n = n0 + (lane & 15);
#pragma unroll
  for (int v = 0; v < 8; ++v)
    out[(size_t)(m0 + v + 8 * half) * (size_t)N + n] = c[v];
}

// ---------------- LayerNorm + ReLU in place over rows of h ------------------
__global__ void k_ln_relu_abce(float* __restrict__ h, const float* __restrict__ g,
                               const float* __restrict__ bt, int H) {
  __shared__ float sd[128];
  int b = blockIdx.x, tid = threadIdx.x;
  float* row = h + (size_t)b * H;
  float s = 0.f;
  for (int i = tid; i < H; i += 128) s += row[i];
  sd[tid] = s; __syncthreads();
  for (int o = 64; o > 0; o >>= 1) { if (tid < o) sd[tid] += sd[tid + o]; __syncthreads(); }
  float mu = sd[0] / (float)H;
  __syncthreads();
  float vv = 0.f;
  for (int i = tid; i < H; i += 128) { float d = row[i] - mu; vv += d * d; }
  sd[tid] = vv; __syncthreads();
  for (int o = 64; o > 0; o >>= 1) { if (tid < o) sd[tid] += sd[tid + o]; __syncthreads(); }
  float rstd = rsqrtf(sd[0] / (float)H + 1e-5f);
  for (int i = tid; i < H; i += 128) {
    float d = (row[i] - mu) * rstd * g[i] + bt[i];
    row[i] = fmaxf(d, 0.f);
  }
}

// --------- fused GEMM + y=0 BCE row-sum (head and tail epilogues) -----------
// Block = 8 waves, all on the same N-tile stream (different M-tiles), so the
// 16xK B tile is staged once per block in LDS as packed-bf16 fragments and the
// per-wave A fragments are hoisted into registers for the whole N loop.
template <int K, bool TAIL>
__global__ void k_fused_bce(const float* __restrict__ A, const float* __restrict__ Bm,
                            const float* __restrict__ root_sig, const float* __restrict__ act,
                            float* __restrict__ rowsum, int Ntiles) {
  constexpr int KS = K / 32;
  __shared__ unsigned btile[KS * 256];       // [kstep][lane][8] packed bf16 pairs
  int lane = threadIdx.x & 31, w = threadIdx.x >> 5;
  int m0 = (blockIdx.y * 8 + w) * 16;
  int half = lane >> 4;

  // hoisted A fragments (constant across the N loop)
  v16bf afr[KS];
#pragma unroll
  for (int s = 0; s < KS; ++s) afr[s] = load_frag(A, K, m0, s * 32, lane);

  float r[8], av[8];
  if constexpr (TAIL) {
#pragma unroll
    for (int v = 0; v < 8; ++v) {
      int m = m0 + v + 8 * half;
      r[v]  = root_sig[m];
      av[v] = act[m];
    }
  }
  float acc[8] = {0.f, 0.f, 0.f, 0.f, 0.f, 0.f, 0.f, 0.f};

  for (int nt = blockIdx.x; nt < Ntiles; nt += gridDim.x) {
    int n0 = nt * 16;
    __syncthreads();                          // protect btile reuse
    // cooperative stage of the 16xK weight tile -> packed bf16 fragment layout
    for (int e = threadIdx.x; e < KS * 256; e += 256) {
      int s  = e >> 8;
      int rr = e & 255;
      int bl = rr >> 3, v = rr & 7;
      int row = n0 + (bl & 15);
      int bk  = frag_bk(v, bl >> 4);
      const float2* p2 = reinterpret_cast<const float2*>(Bm + (size_t)row * K);
      float2 t = p2[(s * 32 + bk) >> 1];
      btile[e] = pack2bf(t.x, t.y);
    }
    // prefetch next tile's rows while this one is consumed
    int ntn = nt + gridDim.x;
    if (ntn < Ntiles) {
      const float* pf = Bm + (size_t)(ntn * 16 + (threadIdx.x & 15)) * K
                        + (size_t)(threadIdx.x >> 4) * (K / 16);
      __builtin_prefetch(pf, 0, 0);
    }
    __syncthreads();

    v8f c = {0.f, 0.f, 0.f, 0.f, 0.f, 0.f, 0.f, 0.f};
#pragma unroll
    for (int s = 0; s < KS; ++s) {
      v8u q = *reinterpret_cast<const v8u*>(&btile[s * 256 + lane * 8]);
      v16bf b = __builtin_bit_cast(v16bf, q);
      c = __builtin_amdgcn_wmma_f32_16x16x32_bf16(false, afr[s], false, b, (short)0, c,
                                                  false, false);
    }
#pragma unroll
    for (int v = 0; v < 8; ++v) {
      if constexpr (TAIL) {
        float sg = 1.f / (1.f + expf(-c[v]));
        float p  = r[v] * sg;
        acc[v] += -fmaxf(log1pf(-p), -100.f);
      } else {
        acc[v] += bce0_logit(c[v]);
      }
    }
  }
#pragma unroll
  for (int v = 0; v < 8; ++v) {
    float s = TAIL ? acc[v] * av[v] : acc[v];
    s += __shfl_xor(s, 1, 32); s += __shfl_xor(s, 2, 32);
    s += __shfl_xor(s, 4, 32); s += __shfl_xor(s, 8, 32);
    if ((lane & 15) == 0) atomicAdd(&rowsum[m0 + v + 8 * half], s);
  }
}

// ---------------- finalize: sparse-label corrections + mean -----------------
__global__ void k_finalize_abce(const float* __restrict__ x, const float* __restrict__ hW,
                                const int* __restrict__ tgt,
                                const float* __restrict__ w2_0, const float* __restrict__ w2_1,
                                const float* __restrict__ w2_2,
                                const float* __restrict__ ws, float* __restrict__ out) {
  __shared__ float red[256];
  int b = threadIdx.x;
  const float* root_raw = ws + OFF_ROOT_RAW;
  const float* root_sig = ws + OFF_ROOT_SIG;
  const float* act      = ws + OFF_ACT;
  const int* tg = tgt + b * 20;
  const float* xr = x + (size_t)b * IN_F;
  float corr = 0.f;
  for (int l = 0; l < 20; ++l) {
    int t = tg[l];
    bool dup = false;
    for (int j = 0; j < l; ++j) if (tg[j] == t) { dup = true; break; }
    if (dup) continue;
    if (t < SHORTN) {
      // head y=1 correction: bce1 - bce0 = -logit
      const float* wr = hW + (size_t)t * IN_F;
      float d = 0.f;
      for (int k = 0; k < IN_F; ++k) d = fmaf(xr[k], wr[k], d);
      corr -= d;
    } else {
      int ci  = (t < 12000) ? 0 : ((t < 40000) ? 1 : 2);
      int low = (ci == 0) ? 2000 : ((ci == 1) ? 12000 : 40000);
      int Kh  = (ci == 0) ? 384  : ((ci == 1) ? 192   : 96);
      const float* hp = (ci == 0) ? (ws + OFF_H0) : ((ci == 1) ? (ws + OFF_H1) : (ws + OFF_H2));
      const float* w2 = (ci == 0) ? w2_0 : ((ci == 1) ? w2_1 : w2_2);
      const float* hr = hp + (size_t)b * Kh;
      const float* wr = w2 + (size_t)(t - low) * Kh;
      float d = 0.f;
      for (int k = 0; k < Kh; ++k) d = fmaf(hr[k], wr[k], d);
      float sg = 1.f / (1.f + expf(-d));
      float p  = root_sig[ci * 256 + b] * sg;
      float c1 = -fmaxf(logf(p), -100.f);      // bce at y=1
      float c0 = -fmaxf(log1pf(-p), -100.f);   // bce at y=0 (already counted)
      corr += act[ci * 256 + b] * (c1 - c0);
    }
  }
  // head cluster columns (SHORT..SHORT+2): contribution = (1-active)*bce0(root_logit)
  float headc = 0.f;
  for (int i = 0; i < 3; ++i) {
    float a = act[i * 256 + b], lg = root_raw[i * 256 + b];
    headc += (1.f - a) * bce0_logit(lg);
  }
  float tot = (ws[OFF_HEADSUM + b] + headc + ws[OFF_CLUSSUM + b] + corr) / ws[OFF_NUML + b];
  red[b] = tot; __syncthreads();
  for (int o = 128; o > 0; o >>= 1) { if (b < o) red[b] += red[b + o]; __syncthreads(); }
  if (b == 0) out[0] = red[0] / 256.f;
}

extern "C" void kernel_launch(void* const* d_in, const int* in_sizes, int n_in,
                              void* d_out, int out_size, void* d_ws, size_t ws_size,
                              hipStream_t stream) {
  (void)in_sizes; (void)n_in; (void)out_size; (void)ws_size;
  const float* x   = (const float*)d_in[0];
  const float* hW  = (const float*)d_in[1];
  const int*   tgt = (const int*)d_in[2];
  const float* w1[3] = {(const float*)d_in[3], (const float*)d_in[7],  (const float*)d_in[11]};
  const float* g [3] = {(const float*)d_in[4], (const float*)d_in[8],  (const float*)d_in[12]};
  const float* bb[3] = {(const float*)d_in[5], (const float*)d_in[9],  (const float*)d_in[13]};
  const float* w2[3] = {(const float*)d_in[6], (const float*)d_in[10], (const float*)d_in[14]};
  float* ws  = (float*)d_ws;
  float* out = (float*)d_out;

  const int    Hs[3]   = {384, 192, 96};
  const int    Os[3]   = {10000, 28000, 60000};
  const size_t hoff[3] = {OFF_H0, OFF_H1, OFF_H2};

  k_prep_abce<<<1, 256, 0, stream>>>(x, hW, tgt, ws);
  for (int i = 0; i < 3; ++i) {
    k_gemm16_abce<<<dim3(Hs[i] / 16, 2), 256, 0, stream>>>(x, w1[i], ws + hoff[i], Hs[i], IN_F);
    k_ln_relu_abce<<<256, 128, 0, stream>>>(ws + hoff[i], g[i], bb[i], Hs[i]);
  }
  // head: 125 N-tiles, one tile per wave
  k_fused_bce<IN_F, false><<<dim3(125, 2), 256, 0, stream>>>(
      x, hW, nullptr, nullptr, ws + OFF_HEADSUM, SHORTN / 16);
  // tails: N-chunked, per-wave register row-accumulators
  k_fused_bce<384, true><<<dim3(256, 2), 256, 0, stream>>>(
      ws + OFF_H0, w2[0], ws + OFF_ROOT_SIG + 0 * 256, ws + OFF_ACT + 0 * 256,
      ws + OFF_CLUSSUM, Os[0] / 16);
  k_fused_bce<192, true><<<dim3(256, 2), 256, 0, stream>>>(
      ws + OFF_H1, w2[1], ws + OFF_ROOT_SIG + 1 * 256, ws + OFF_ACT + 1 * 256,
      ws + OFF_CLUSSUM, Os[1] / 16);
  k_fused_bce<96, true><<<dim3(256, 2), 256, 0, stream>>>(
      ws + OFF_H2, w2[2], ws + OFF_ROOT_SIG + 2 * 256, ws + OFF_ACT + 2 * 256,
      ws + OFF_CLUSSUM, Os[2] / 16);
  k_finalize_abce<<<1, 256, 0, stream>>>(x, hW, tgt, w2[0], w2[1], w2[2], ws, out);
}
